// StructureEncoder_75402445848858
// MI455X (gfx1250) — matrix operand order repaired and checked by reading, not compiled, not run on hardware
//
#include <hip/hip_runtime.h>

typedef _Float16 v16h __attribute__((ext_vector_type(16)));
typedef _Float16 v8h  __attribute__((ext_vector_type(8)));
typedef float    v8f  __attribute__((ext_vector_type(8)));

#define LSZ   512
#define KREP  4
#define D0    128
#define CIN   196
#define CPAD  224          // per-tap K padded to 7 chunks of 32
#define KCH   7
#define TILE_X 32
#define HALO_W 34
#define LDS_ELEMS (3*HALO_W*CPAD)

// ---------------- quaternion helpers (reference conventions) ----------------
struct Qt { float w, x, y, z; };

__device__ inline Qt qmm(Qt a, Qt b) {
    // a = q1, b = q2 ; vector part = cross(q2_v, q1_v) + q1.w*q2_v + q2.w*q1_v
    Qt r;
    r.w = a.w*b.w - (a.x*b.x + a.y*b.y + a.z*b.z);
    r.x = (b.y*a.z - b.z*a.y) + a.w*b.x + b.w*a.x;
    r.y = (b.z*a.x - b.x*a.z) + a.w*b.y + b.w*a.y;
    r.z = (b.x*a.y - b.y*a.x) + a.w*b.z + b.w*a.z;
    return r;
}

__device__ inline void trot(Qt q, float tx, float ty, float tz,
                            float& ox, float& oy, float& oz) {
    Qt t4{0.f, tx, ty, tz};
    Qt qi{q.w, -q.x, -q.y, -q.z};
    Qt r = qmm(qmm(q, t4), qi);
    ox = r.x; oy = r.y; oz = r.z;
}

__device__ inline float sigmoidf(float v) { return 1.f / (1.f + __expf(-v)); }

// ---------------- feature construction: writes f16 X[196][L][L] -------------
__global__ __launch_bounds__(256)
void feat_kernel(const float* __restrict__ x2D,
                 const float* __restrict__ quat,   // [K][L][4]
                 const float* __restrict__ trans,  // [K][L][3]
                 _Float16* __restrict__ X)         // [CIN][L][L]
{
    const size_t LL = (size_t)LSZ * LSZ;
    size_t idx = (size_t)blockIdx.x * blockDim.x + threadIdx.x;
    if (idx >= LL) return;
    int i = (int)(idx / LSZ);   // l1
    int j = (int)(idx % LSZ);   // l2

    // channels 0..127 : x2D cast to f16
    for (int c = 0; c < D0; ++c)
        X[c*LL + idx] = (_Float16)x2D[c*LL + idx];

    const float tcb_x = -0.537f, tcb_y = -0.769f, tcb_z = -1.208f; // C_N_CB[:,2]

    float distCA = 0.f, distCB = 0.f;
    float tax = 0.f, tay = 0.f, taz = 0.f;
    float qaw = 0.f, qax = 0.f, qay = 0.f, qaz = 0.f;

    for (int k = 0; k < KREP; ++k) {
        const float* qpi = quat  + ((size_t)k*LSZ + i)*4;
        const float* qpj = quat  + ((size_t)k*LSZ + j)*4;
        const float* tpi = trans + ((size_t)k*LSZ + i)*3;
        const float* tpj = trans + ((size_t)k*LSZ + j)*3;
        Qt qi{qpi[0], qpi[1], qpi[2], qpi[3]};
        Qt qj{qpj[0], qpj[1], qpj[2], qpj[3]};
        float tix = tpi[0], tiy = tpi[1], tiz = tpi[2];
        float tjx = tpj[0], tjy = tpj[1], tjz = tpj[2];

        // CA affinity distance
        float dx = tjx - tix, dy = tjy - tiy, dz = tjz - tiz;
        distCA += sqrtf(dx*dx + dy*dy + dz*dz + 1e-8f);

        // CB coords: trot(q, tcb) + t
        float ax, ay, az, bx, by, bz;
        trot(qi, tcb_x, tcb_y, tcb_z, ax, ay, az);
        trot(qj, tcb_x, tcb_y, tcb_z, bx, by, bz);
        float cx = (bx + tjx) - (ax + tix);
        float cy = (by + tjy) - (ay + tiy);
        float cz = (bz + tjz) - (az + tiz);
        distCB += sqrtf(cx*cx + cy*cy + cz*cz + 1e-8f);

        // tL : rotate (tj - ti) by conj(qi)
        Qt qic{qi.w, -qi.x, -qi.y, -qi.z};
        float rx, ry, rz;
        trot(qic, dx, dy, dz, rx, ry, rz);
        tax += rx; tay += ry; taz += rz;

        // qL : norm_quat(qmm(conj(qi), qj))
        Qt qr = qmm(qic, qj);
        float nrm = rsqrtf(qr.w*qr.w + qr.x*qr.x + qr.y*qr.y + qr.z*qr.z + 1e-6f);
        float sgn = (qr.w >= 0.f) ? 1.f : -1.f;
        float s = nrm * sgn;
        qaw += qr.w*s; qax += qr.x*s; qay += qr.y*s; qaz += qr.z*s;
    }
    distCA *= (1.f/KREP); distCB *= (1.f/KREP);
    tax *= (1.f/KREP); tay *= (1.f/KREP); taz *= (1.f/KREP);
    qaw *= (1.f/KREP); qax *= (1.f/KREP); qay *= (1.f/KREP); qaz *= (1.f/KREP);

    // aff_CB -> channels 128..144 ; aff_CA -> 145..161 (cutoffs 8,10,...,40)
    for (int b = 0; b < 17; ++b) {
        float cut = (float)(4 + b) * 2.f;
        X[(size_t)(128 + b)*LL + idx] = (_Float16)sigmoidf(distCB - cut);
        X[(size_t)(145 + b)*LL + idx] = (_Float16)sigmoidf(distCA - cut);
    }
    // tL -> channels 162..191 : sigmoid(t[d] / ((c+1)*10))
    float tv[3] = {tax, tay, taz};
    for (int d = 0; d < 3; ++d)
        for (int c = 0; c < 10; ++c)
            X[(size_t)(162 + d*10 + c)*LL + idx] =
                (_Float16)sigmoidf(tv[d] * (1.f / ((float)(c + 1) * 10.f)));
    // qL -> channels 192..195 : norm_quat(mean)
    float nrm = rsqrtf(qaw*qaw + qax*qax + qay*qay + qaz*qaz + 1e-6f);
    float sgn = (qaw >= 0.f) ? 1.f : -1.f;
    float s = nrm * sgn;
    X[(size_t)192*LL + idx] = (_Float16)(qaw*s);
    X[(size_t)193*LL + idx] = (_Float16)(qax*s);
    X[(size_t)194*LL + idx] = (_Float16)(qay*s);
    X[(size_t)195*LL + idx] = (_Float16)(qaz*s);
}

// ---------------- per-channel instance-norm stats ---------------------------
__global__ __launch_bounds__(256)
void stats_kernel(const _Float16* __restrict__ X, float* __restrict__ mean,
                  float* __restrict__ rstd)
{
    const size_t LL = (size_t)LSZ * LSZ;
    int c = blockIdx.x;
    const _Float16* p = X + (size_t)c * LL;
    float s = 0.f, sq = 0.f;
    for (size_t t = threadIdx.x; t < LL; t += 256) {
        float v = (float)p[t];
        s += v; sq += v * v;
    }
    __shared__ float sh1[256], sh2[256];
    sh1[threadIdx.x] = s; sh2[threadIdx.x] = sq;
    __syncthreads();
    for (int o = 128; o > 0; o >>= 1) {
        if (threadIdx.x < o) {
            sh1[threadIdx.x] += sh1[threadIdx.x + o];
            sh2[threadIdx.x] += sh2[threadIdx.x + o];
        }
        __syncthreads();
    }
    if (threadIdx.x == 0) {
        float m = sh1[0] / (float)LL;
        float v = sh2[0] / (float)LL - m * m;
        mean[c] = m;
        rstd[c] = rsqrtf(v + 1e-5f);
    }
}

// ---------------- weight pack into WMMA B-fragment layout -------------------
// wp layout: [nTiles][9 taps][KCH][32 lanes][16 f16]
// B 32x16 f16 fragment: lane = n + 16*h ; element e -> K = 8*h + e (e<8), 16 + 8*h + (e-8) (e>=8)
__global__ __launch_bounds__(256)
void pack_kernel(const float* __restrict__ w, _Float16* __restrict__ wp,
                 int Cout, int total)
{
    int idx = blockIdx.x * blockDim.x + threadIdx.x;
    if (idx >= total) return;
    int e    = idx & 15;
    int rest = idx >> 4;
    int lane = rest & 31;  rest >>= 5;
    int kk   = rest % KCH; rest /= KCH;
    int t    = rest % 9;
    int h  = lane >> 4;
    int nl = lane & 15;
    int nt = rest / 9;
    int n  = nt*16 + nl;
    int c  = kk*32 + ((e < 8) ? (8*h + e) : (16 + 8*h + (e - 8)));
    int r  = t / 3, s = t % 3;
    float v = 0.f;
    if (n < Cout && c < CIN)
        v = w[(((size_t)n*CIN + c)*3 + r)*3 + s];
    wp[idx] = (_Float16)v;
}

// ---------------- implicit-GEMM 3x3 conv via WMMA ---------------------------
__global__ __launch_bounds__(256)
void conv_wmma_kernel(const _Float16* __restrict__ Xin,
                      const float* __restrict__ mean,
                      const float* __restrict__ rstd,
                      const _Float16* __restrict__ Wp,
                      const float* __restrict__ bias,
                      const float* __restrict__ resid,   // x2D or nullptr
                      _Float16* __restrict__ outH,       // f16 act out (leaky) or nullptr
                      float* __restrict__ outF,          // f32 out (residual) or nullptr
                      int Cout, int nTiles)
{
    __shared__ __align__(16) _Float16 smem[LDS_ELEMS];
    const size_t LL = (size_t)LSZ * LSZ;
    const int tid = threadIdx.x;
    const int seg = blockIdx.x & 15;
    const int y   = blockIdx.x >> 4;
    const int x0  = seg * TILE_X;

    // stage 3x34xCPAD tile, normalization folded in, halo & pad channels = 0
    for (int idx = tid; idx < LDS_ELEMS; idx += 256) {
        int c   = idx / (3 * HALO_W);
        int rem = idx - c * (3 * HALO_W);
        int rr  = rem / HALO_W;
        int xi  = rem - rr * HALO_W;
        int yin = y + rr - 1;
        int xin = x0 + xi - 1;
        float v = 0.f;
        if (c < CIN && yin >= 0 && yin < LSZ && xin >= 0 && xin < LSZ)
            v = ((float)Xin[(size_t)c*LL + (size_t)yin*LSZ + xin] - mean[c]) * rstd[c];
        smem[(rr*HALO_W + xi)*CPAD + c] = (_Float16)v;
    }
    __syncthreads();

    const int wid  = tid >> 5;
    const int lane = tid & 31;
    const int mt    = wid & 1;                       // pixel sub-tile 0/1
    const int ntile = (blockIdx.y << 2) + (wid >> 1);
    if (ntile >= nTiles) return;                     // wave-uniform exit

    const int h  = lane >> 4;
    const int nl = lane & 15;
    const int arow = nl + mt*16;                     // A-matrix row (pixel in segment)

    v8f acc = {0.f,0.f,0.f,0.f,0.f,0.f,0.f,0.f};
    const _Float16* wpb = Wp + (size_t)ntile * 9 * KCH * 512;

    for (int t = 0; t < 9; ++t) {
        const int rr = t / 3, ss = t - rr*3;
        const _Float16* ap = &smem[(rr*HALO_W + (arow + ss)) * CPAD];
        #pragma unroll
        for (int kk = 0; kk < KCH; ++kk) {
            // A fragment: lane half h reads channels {8h..8h+7} and {16+8h..23+8h}
            const int cb = kk*32 + 8*h;
            v8h alo = *(const v8h*)(ap + cb);
            v8h ahi = *(const v8h*)(ap + cb + 16);
            v16h A;
            #pragma unroll
            for (int e = 0; e < 8; ++e) { A[e] = alo[e]; A[e+8] = ahi[e]; }
            // B fragment: 32 contiguous bytes per lane (pre-packed)
            const _Float16* bp = wpb + ((size_t)(t*KCH + kk)*32 + lane)*16;
            v8h blo = *(const v8h*)(bp);
            v8h bhi = *(const v8h*)(bp + 8);
            v16h Bv;
            #pragma unroll
            for (int e = 0; e < 8; ++e) { Bv[e] = blo[e]; Bv[e+8] = bhi[e]; }
            acc = __builtin_amdgcn_wmma_f32_16x16x32_f16(
                      false, A, false, Bv, (short)0, acc, false, false);
        }
    }

    // epilogue: D element r -> pixel row r + 8*h ; column nl -> out channel
    const int n = ntile*16 + nl;
    if (n < Cout) {
        const float b = bias[n];
        #pragma unroll
        for (int r = 0; r < 8; ++r) {
            const int mrow = r + 8*h;
            const int xpix = x0 + mt*16 + mrow;
            float v = acc[r] + b;
            const size_t off = (size_t)n*LL + (size_t)y*LSZ + xpix;
            if (outH) {
                v = (v > 0.f) ? v : 0.01f * v;       // leaky relu
                outH[off] = (_Float16)v;
            } else {
                outF[off] = resid[off] + v;          // x2D residual
            }
        }
    }
}

// ---------------- launch ----------------------------------------------------
extern "C" void kernel_launch(void* const* d_in, const int* in_sizes, int n_in,
                              void* d_out, int out_size, void* d_ws, size_t ws_size,
                              hipStream_t stream) {
    const float* x2D     = (const float*)d_in[0];
    const float* quat    = (const float*)d_in[1];
    const float* trans   = (const float*)d_in[2];
    const float* conv1_w = (const float*)d_in[3];
    const float* conv1_b = (const float*)d_in[4];
    const float* conv2_w = (const float*)d_in[5];
    const float* conv2_b = (const float*)d_in[6];
    const float* conv3_w = (const float*)d_in[7];
    const float* conv3_b = (const float*)d_in[8];
    float* out = (float*)d_out;

    const size_t LL = (size_t)LSZ * LSZ;
    _Float16* X0 = (_Float16*)d_ws;                 // features / act2 (reused)
    _Float16* A1 = X0 + (size_t)CIN * LL;           // act1
    float* stats = (float*)(A1 + (size_t)CIN * LL);
    float* mean0 = stats;        float* rstd0 = stats + 256;
    float* mean1 = stats + 512;  float* rstd1 = stats + 768;
    float* mean2 = stats + 1024; float* rstd2 = stats + 1280;
    _Float16* Wp1 = (_Float16*)(stats + 1536);
    const size_t wpsz12 = (size_t)13 * 9 * KCH * 512;   // 13 n-tiles
    const size_t wpsz3  = (size_t)8  * 9 * KCH * 512;   // 8 n-tiles
    _Float16* Wp2 = Wp1 + wpsz12;
    _Float16* Wp3 = Wp2 + wpsz12;

    // 1. features (+ x2D cast) -> X0 f16
    feat_kernel<<<dim3((unsigned)(LL / 256)), 256, 0, stream>>>(x2D, quat, trans, X0);

    // 2. pack weights into WMMA B layout
    pack_kernel<<<dim3((unsigned)((wpsz12 + 255) / 256)), 256, 0, stream>>>(
        conv1_w, Wp1, CIN, (int)wpsz12);
    pack_kernel<<<dim3((unsigned)((wpsz12 + 255) / 256)), 256, 0, stream>>>(
        conv2_w, Wp2, CIN, (int)wpsz12);
    pack_kernel<<<dim3((unsigned)((wpsz3 + 255) / 256)), 256, 0, stream>>>(
        conv3_w, Wp3, D0, (int)wpsz3);

    const dim3 cgrid(16 * LSZ, 4);   // 32-pixel segments x 16 n-tiles (13 used)
    const dim3 cgrid3(16 * LSZ, 2);  // 8 n-tiles

    // 3. IN + conv1 + leaky -> A1
    stats_kernel<<<CIN, 256, 0, stream>>>(X0, mean0, rstd0);
    conv_wmma_kernel<<<cgrid, 256, 0, stream>>>(X0, mean0, rstd0, Wp1, conv1_b,
                                                nullptr, A1, nullptr, CIN, 13);
    // 4. IN + conv2 + leaky -> X0 (reuse)
    stats_kernel<<<CIN, 256, 0, stream>>>(A1, mean1, rstd1);
    conv_wmma_kernel<<<cgrid, 256, 0, stream>>>(A1, mean1, rstd1, Wp2, conv2_b,
                                                nullptr, X0, nullptr, CIN, 13);
    // 5. IN + conv3 + residual -> d_out f32
    stats_kernel<<<CIN, 256, 0, stream>>>(X0, mean2, rstd2);
    conv_wmma_kernel<<<cgrid3, 256, 0, stream>>>(X0, mean2, rstd2, Wp3, conv3_b,
                                                 x2D, nullptr, out, D0, 8);
}
